// FrameWiseHGNN_21294447854181
// MI455X (gfx1250) — compile-verified, matrix operand order checked by reading, not run on hardware
//
#include <hip/hip_runtime.h>
#include <math.h>

// ---------------------------------------------------------------------------
// FrameWiseHGNN for MI455X (gfx1250, wave32, WMMA)
//
// Pipeline (per batch b of 4, N=2048 nodes, H1=128, C=8):
//   1. knn_kernel        : 1 wave per center -> top-3 sq-dists, avg dist,
//                          incidence vals = exp(-d2/avg^2), edge degree de.
//   2. scatter_dv        : vertex degrees dv via atomicAdd.
//   3. transpose_w1      : W1 [2048,128] f32 -> W1t [128,2048] f16.
//   4. per batch:
//        zero_g          : clear dense f32 G (2048x2048).
//        scatter_g       : each edge adds a 3x3 block vals_j*vals_k/de.
//        scale_convert_g : G16 = f16( G32 * dv^-1/2 outer ).
//        gemm1_wmma      : F1t = (G @ W1 + b1)^T  (f16, v_wmma_f32_16x16x32_f16)
//        gemm2_wmma      : rowmax(G @ F1) -> encoded atomicMax pool
//   5. classify          : out = maxpool @ Wc + bc.
// ---------------------------------------------------------------------------

typedef __attribute__((ext_vector_type(16))) _Float16 v16h;
typedef __attribute__((ext_vector_type(8)))  _Float16 v8h;
typedef __attribute__((ext_vector_type(8)))  float    v8f;

#define N_NODES 2048
#define BATCH   4
#define H1DIM   128
#define NCLS    8

// monotone float <-> uint mapping so atomicMax(unsigned) == float max
__device__ __forceinline__ unsigned fenc(float f) {
  unsigned u = __float_as_uint(f);
  return (u & 0x80000000u) ? ~u : (u | 0x80000000u);
}
__device__ __forceinline__ float fdec(unsigned u) {
  unsigned v = (u & 0x80000000u) ? (u ^ 0x80000000u) : ~u;
  return __uint_as_float(v);
}

// --------------------------------------------------------------------------
// 1 wave32 per (batch, center): scan all 2048 nodes, keep 3 smallest sq-dists
// and sum of sqrt(d2+eps); merge per-lane top3 in LDS; emit idx/vals/de.
// --------------------------------------------------------------------------
__global__ __launch_bounds__(256) void knn_kernel(const float* __restrict__ x,
                                                  int* __restrict__ nidx,
                                                  float* __restrict__ nval,
                                                  float* __restrict__ de) {
  __shared__ float sd[8][96];
  __shared__ int   si[8][96];
  int wave = (blockIdx.x * blockDim.x + threadIdx.x) >> 5;   // 0..8191
  int lane = threadIdx.x & 31;
  int ws   = threadIdx.x >> 5;
  int b = wave >> 11;
  int c = wave & (N_NODES - 1);
  const float* xb = x + (size_t)b * N_NODES * 3;
  float cx = xb[c * 3 + 0], cy = xb[c * 3 + 1], cz = xb[c * 3 + 2];
  float b0 = 3.4e38f, b1v = 3.4e38f, b2 = 3.4e38f;
  int   i0 = 0, i1 = 0, i2 = 0;
  float dsum = 0.f;
  for (int j = lane; j < N_NODES; j += 32) {
    float dx = cx - xb[j * 3 + 0];
    float dy = cy - xb[j * 3 + 1];
    float dz = cz - xb[j * 3 + 2];
    float d2 = dx * dx + dy * dy + dz * dz;
    dsum += sqrtf(d2 + 1e-12f);
    if (d2 < b2) {
      if (d2 < b1v) {
        b2 = b1v; i2 = i1;
        if (d2 < b0) { b1v = b0; i1 = i0; b0 = d2; i0 = j; }
        else         { b1v = d2; i1 = j; }
      } else { b2 = d2; i2 = j; }
    }
  }
  for (int m = 16; m > 0; m >>= 1) dsum += __shfl_xor(dsum, m, 32);
  sd[ws][lane * 3 + 0] = b0;  si[ws][lane * 3 + 0] = i0;
  sd[ws][lane * 3 + 1] = b1v; si[ws][lane * 3 + 1] = i1;
  sd[ws][lane * 3 + 2] = b2;  si[ws][lane * 3 + 2] = i2;
  __syncthreads();
  if (lane == 0) {
    float avg = dsum * (1.0f / N_NODES);
    float inv = 1.0f / (avg * avg);
    float s = 0.f;
    for (int k = 0; k < 3; ++k) {
      float best = 3.5e38f; int bj = 0, bp = 0;
      for (int p = 0; p < 96; ++p) {
        float v = sd[ws][p];
        if (v < best) { best = v; bj = si[ws][p]; bp = p; }
      }
      sd[ws][bp] = 3.5e38f;
      float val = expf(-best * inv);
      nidx[(size_t)wave * 3 + k] = bj;
      nval[(size_t)wave * 3 + k] = val;
      s += val;
    }
    de[wave] = s;
  }
}

__global__ void init_small(float* dv, unsigned* pool) {
  int i = blockIdx.x * blockDim.x + threadIdx.x;
  if (i < BATCH * N_NODES) dv[i] = 0.f;
  if (i < BATCH * H1DIM)   pool[i] = 0u;   // 0 encodes below -FLT_MAX
}

__global__ void scatter_dv(const int* __restrict__ nidx,
                           const float* __restrict__ nval,
                           float* __restrict__ dv) {
  int t = blockIdx.x * blockDim.x + threadIdx.x;   // (b, edge)
  if (t >= BATCH * N_NODES) return;
  int b = t >> 11;
  for (int k = 0; k < 3; ++k)
    atomicAdd(dv + b * N_NODES + nidx[(size_t)t * 3 + k], nval[(size_t)t * 3 + k]);
}

__global__ void transpose_w1(const float* __restrict__ W1, _Float16* __restrict__ W1t) {
  int i = blockIdx.x * blockDim.x + threadIdx.x;   // 128*2048
  if (i >= N_NODES * H1DIM) return;
  int k = i & (N_NODES - 1);
  int n = i >> 11;
  W1t[(size_t)n * N_NODES + k] = (_Float16)W1[(size_t)k * H1DIM + n];
}

__global__ void zero_g(float* __restrict__ G) {
  size_t i = (size_t)blockIdx.x * blockDim.x + threadIdx.x;
  size_t stride = (size_t)gridDim.x * blockDim.x;
  for (; i < (size_t)N_NODES * N_NODES; i += stride) G[i] = 0.f;
}

__global__ void scatter_g(const int* __restrict__ nidx,
                          const float* __restrict__ nval,
                          const float* __restrict__ de,
                          float* __restrict__ G, int b) {
  int e = blockIdx.x * blockDim.x + threadIdx.x;
  if (e >= N_NODES) return;
  size_t base = ((size_t)b * N_NODES + e) * 3;
  int   ii[3] = { nidx[base], nidx[base + 1], nidx[base + 2] };
  float vv[3] = { nval[base], nval[base + 1], nval[base + 2] };
  float invde = 1.0f / de[b * N_NODES + e];
  for (int j = 0; j < 3; ++j)
    for (int k = 0; k < 3; ++k)
      atomicAdd(&G[(size_t)ii[j] * N_NODES + ii[k]], vv[j] * vv[k] * invde);
}

__global__ void scale_convert_g(const float* __restrict__ G32,
                                const float* __restrict__ dv,
                                _Float16* __restrict__ G16, int b) {
  size_t i = (size_t)blockIdx.x * blockDim.x + threadIdx.x;
  if (i >= (size_t)N_NODES * N_NODES) return;
  int r = (int)(i >> 11);
  int c = (int)(i & (N_NODES - 1));
  float g = G32[i] * rsqrtf(dv[b * N_NODES + r]) * rsqrtf(dv[b * N_NODES + c]);
  G16[i] = (_Float16)g;
}

// --------------------------------------------------------------------------
// WMMA GEMM: one wave per 16x16 tile of C = G[2048x2048] x B[2048x128].
// Fragment addressing per CDNA5 16-bit layout: lane l -> row (l&15),
// K halves base..base+7 and base+16..base+23 with base=(l>>4)*8.
// B operand supplied transposed so it uses the identical contiguous pattern.
// --------------------------------------------------------------------------
__global__ __launch_bounds__(256) void gemm1_wmma(const _Float16* __restrict__ G,
                                                  const _Float16* __restrict__ W1t,
                                                  const float* __restrict__ b1,
                                                  _Float16* __restrict__ F1t) {
  int wave = threadIdx.x >> 5;
  int tile = blockIdx.x * 8 + wave;        // 1024 tiles: 128 M-tiles x 8 N-tiles
  int tM = tile >> 3, tN = tile & 7;
  int lane = threadIdx.x & 31;
  int r = lane & 15, hs = lane >> 4;
  const _Float16* ar = G   + ((size_t)(tM * 16 + r)) * N_NODES + hs * 8;
  const _Float16* br = W1t + ((size_t)(tN * 16 + r)) * N_NODES + hs * 8;
  v8f c = {};
  for (int k0 = 0; k0 < N_NODES; k0 += 32) {
    v8h alo = *(const v8h*)(ar + k0);
    v8h ahi = *(const v8h*)(ar + k0 + 16);
    v8h blo = *(const v8h*)(br + k0);
    v8h bhi = *(const v8h*)(br + k0 + 16);
    v16h a, bm;
#pragma unroll
    for (int i = 0; i < 8; ++i) {
      a[i] = alo[i];  a[i + 8] = ahi[i];
      bm[i] = blo[i]; bm[i + 8] = bhi[i];
    }
    c = __builtin_amdgcn_wmma_f32_16x16x32_f16(false, a, false, bm, (short)0, c,
                                               false, false);
  }
  // C/D layout: VGPR g holds (m = g + 8*hs, n = r). Store transposed + bias.
  float bias = b1[tN * 16 + r];
  _Float16* dst = F1t + ((size_t)(tN * 16 + r)) * N_NODES + tM * 16 + hs * 8;
#pragma unroll
  for (int g = 0; g < 8; ++g) dst[g] = (_Float16)(c[g] + bias);
}

__global__ __launch_bounds__(256) void gemm2_wmma(const _Float16* __restrict__ G,
                                                  const _Float16* __restrict__ F1t,
                                                  unsigned* __restrict__ pool) {
  int wave = threadIdx.x >> 5;
  int tile = blockIdx.x * 8 + wave;
  int tM = tile >> 3, tN = tile & 7;
  int lane = threadIdx.x & 31;
  int r = lane & 15, hs = lane >> 4;
  const _Float16* ar = G   + ((size_t)(tM * 16 + r)) * N_NODES + hs * 8;
  const _Float16* br = F1t + ((size_t)(tN * 16 + r)) * N_NODES + hs * 8;
  v8f c = {};
  for (int k0 = 0; k0 < N_NODES; k0 += 32) {
    v8h alo = *(const v8h*)(ar + k0);
    v8h ahi = *(const v8h*)(ar + k0 + 16);
    v8h blo = *(const v8h*)(br + k0);
    v8h bhi = *(const v8h*)(br + k0 + 16);
    v16h a, bm;
#pragma unroll
    for (int i = 0; i < 8; ++i) {
      a[i] = alo[i];  a[i + 8] = ahi[i];
      bm[i] = blo[i]; bm[i + 8] = bhi[i];
    }
    c = __builtin_amdgcn_wmma_f32_16x16x32_f16(false, a, false, bm, (short)0, c,
                                               false, false);
  }
  float m = c[0];
#pragma unroll
  for (int g = 1; g < 8; ++g) m = fmaxf(m, c[g]);
  atomicMax(pool + tN * 16 + r, fenc(m));   // max over node dimension
}

__global__ void classify(const unsigned* __restrict__ pool,
                         const float* __restrict__ Wc,
                         const float* __restrict__ bc,
                         float* __restrict__ out) {
  int t = threadIdx.x;
  if (t >= BATCH * NCLS) return;
  int b = t >> 3, c = t & 7;
  float s = bc[c];
  for (int h = 0; h < H1DIM; ++h)
    s += fdec(pool[b * H1DIM + h]) * Wc[h * NCLS + c];
  out[t] = s;
}

extern "C" void kernel_launch(void* const* d_in, const int* in_sizes, int n_in,
                              void* d_out, int out_size, void* d_ws, size_t ws_size,
                              hipStream_t stream) {
  const float* x  = (const float*)d_in[0];   // [4,2048,3]
  const float* W1 = (const float*)d_in[1];   // [2048,128]
  const float* b1 = (const float*)d_in[2];   // [128]
  const float* Wc = (const float*)d_in[3];   // [128,8]
  const float* bc = (const float*)d_in[4];   // [8]
  float* out = (float*)d_out;                // [4,8]

  // Workspace carve-out (~28 MB total, 256B aligned blocks)
  char* w = (char*)d_ws;
  size_t off = 0;
  auto carve = [&](size_t bytes) {
    char* p = w + off;
    off = (off + bytes + 255) & ~(size_t)255;
    return p;
  };
  float*    G32  = (float*)carve((size_t)N_NODES * N_NODES * 4);        // 16 MB (per-batch reuse)
  _Float16* G16  = (_Float16*)carve((size_t)N_NODES * N_NODES * 2);     // 8 MB (per-batch reuse)
  _Float16* W1t  = (_Float16*)carve((size_t)H1DIM * N_NODES * 2);       // 512 KB
  _Float16* F1t  = (_Float16*)carve((size_t)BATCH * H1DIM * N_NODES * 2); // 2 MB
  int*      nidx = (int*)carve((size_t)BATCH * N_NODES * 3 * 4);
  float*    nval = (float*)carve((size_t)BATCH * N_NODES * 3 * 4);
  float*    de   = (float*)carve((size_t)BATCH * N_NODES * 4);
  float*    dv   = (float*)carve((size_t)BATCH * N_NODES * 4);
  unsigned* pool = (unsigned*)carve((size_t)BATCH * H1DIM * 4);

  knn_kernel<<<1024, 256, 0, stream>>>(x, nidx, nval, de);
  init_small<<<32, 256, 0, stream>>>(dv, pool);
  scatter_dv<<<32, 256, 0, stream>>>(nidx, nval, dv);
  transpose_w1<<<1024, 256, 0, stream>>>(W1, W1t);

  for (int b = 0; b < BATCH; ++b) {
    _Float16* F1t_b = F1t + (size_t)b * H1DIM * N_NODES;
    zero_g<<<2048, 256, 0, stream>>>(G32);
    scatter_g<<<8, 256, 0, stream>>>(nidx, nval, de, G32, b);
    scale_convert_g<<<16384, 256, 0, stream>>>(G32, dv, G16, b);
    gemm1_wmma<<<128, 256, 0, stream>>>(G16, W1t, b1, F1t_b);
    gemm2_wmma<<<128, 256, 0, stream>>>(G16, F1t_b, pool + b * H1DIM);
  }
  classify<<<1, 32, 0, stream>>>(pool, Wc, bc, out);
}
